// NonLocal_44736379355383
// MI455X (gfx1250) — matrix operand order repaired and checked by reading, not compiled, not run on hardware
//
#include <hip/hip_runtime.h>

// ---------------------------------------------------------------------------
// Non-local attention block for MI455X (gfx1250), bf16 WMMA with vectorized
// operand feeds (all WMMA A/B loads are contiguous b128s).
// ---------------------------------------------------------------------------

typedef __attribute__((ext_vector_type(16))) __bf16 v16bf;
typedef __attribute__((ext_vector_type(8)))  __bf16 v8bf;
typedef __attribute__((ext_vector_type(8)))  float  v8f;

#define N_    8
#define C_    512
#define C8_   64
#define C2_   256
#define H_    64
#define W_    64
#define HW_   4096
#define HW4_  1024
#define MPROJ_ 384   // 64 theta + 64 phi + 256 g stacked output channels

// --- WMMA 16x16x32 bf16 lane/element layouts (CDNA5 ISA) ------------
// A (16x32 MxK): lane row = lane&15; lanes 0-15 hold K {0..7,16..23},
//                lanes 16-31 hold K {8..15,24..31}  -> two contiguous 8-runs.
// B (32x16 KxN): lane col = lane&15; same K split as A:
//                lanes 0-15 K {0..7,16..23}, lanes 16-31 K {8..15,24..31}.
// C/D (16x16 f32): VGPR r -> row M = (lane>>4)*8 + r, col N = lane&15.

__device__ __forceinline__ v16bf load_a16(const __bf16* p, int lane) {
  const int kb = (lane >> 4) << 3;                 // 0 or 8
  const v8bf lo = *(const v8bf*)(p + kb);          // K = kb .. kb+7
  const v8bf hi = *(const v8bf*)(p + kb + 16);     // K = kb+16 .. kb+23
  v16bf r;
  #pragma unroll
  for (int i = 0; i < 8; ++i) { r[i] = lo[i]; r[i + 8] = hi[i]; }
  return r;
}
__device__ __forceinline__ v16bf load_b16(const __bf16* p, int lane) {
  const int kb = (lane >> 4) << 3;                 // 0 or 8
  const v8bf lo = *(const v8bf*)(p + kb);          // K = kb .. kb+7
  const v8bf hi = *(const v8bf*)(p + kb + 16);     // K = kb+16 .. kb+23
  v16bf r;
  #pragma unroll
  for (int i = 0; i < 8; ++i) { r[i] = lo[i]; r[i + 8] = hi[i]; }
  return r;
}
__device__ __forceinline__ v8f wmma_bf16(v16bf a, v16bf b, v8f c) {
  return __builtin_amdgcn_wmma_f32_16x16x32_bf16(
      false, a, false, b, (short)0, c, false, false);
}

// ---------------------------------------------------------------------------
// Kernel 0a: convert stacked weights to bf16.
//   wbf  [384][512] : rows 0-63 theta_w, 64-127 phi_w, 128-383 g_w
//   owbf [512][256] : out_w
// ---------------------------------------------------------------------------
__global__ void k_cvt_w(const float* __restrict__ tw, const float* __restrict__ pw,
                        const float* __restrict__ gw, const float* __restrict__ ow,
                        __bf16* __restrict__ wbf, __bf16* __restrict__ owbf)
{
  const int idx = blockIdx.x * blockDim.x + threadIdx.x;
  if (idx < MPROJ_ * C_) {
    const int k = idx >> 9;
    const int c = idx & 511;
    const float v = (k < 64) ? tw[k * C_ + c]
                  : (k < 128) ? pw[(k - 64) * C_ + c] : gw[(k - 128) * C_ + c];
    wbf[idx] = (__bf16)v;
  } else {
    const int i2 = idx - MPROJ_ * C_;
    if (i2 < C_ * C2_) owbf[i2] = (__bf16)ow[i2];
  }
}

// ---------------------------------------------------------------------------
// Kernel 0b: x [N,C,HW] f32 -> xT [N,HW,C] bf16 (LDS-tiled transpose).
// ---------------------------------------------------------------------------
__global__ __launch_bounds__(256) void k_xpose(const float* __restrict__ x,
                                               __bf16* __restrict__ xT)
{
  __shared__ float t[32][33];
  const int n   = blockIdx.z;
  const int hw0 = blockIdx.x * 32;
  const int c0  = blockIdx.y * 32;
  const int tx  = threadIdx.x & 31;
  const int ty  = threadIdx.x >> 5;          // 0..7
  #pragma unroll
  for (int i = 0; i < 4; ++i) {
    const int c = ty + i * 8;
    t[c][tx] = x[((size_t)n * C_ + c0 + c) * HW_ + hw0 + tx];
  }
  __syncthreads();
  #pragma unroll
  for (int i = 0; i < 4; ++i) {
    const int hw = ty + i * 8;
    xT[((size_t)n * HW_ + hw0 + hw) * C_ + c0 + tx] = (__bf16)t[tx][hw];
  }
}

// ---------------------------------------------------------------------------
// Kernel 1: fused projection GEMM. proj[n][k][hw] = W[k,:] . x[n,:,hw] + b[k]
// M=384, N=4096, K=512.  WG=128thr/4 waves; WG tile 64k x 64hw; wave 16x64.
// ---------------------------------------------------------------------------
__global__ __launch_bounds__(128) void k_proj(
    const __bf16* __restrict__ xT,     // [N][HW][C] bf16
    const __bf16* __restrict__ wbf,    // [384][512] bf16
    const float* __restrict__ theta_b, const float* __restrict__ phi_b,
    const float* __restrict__ g_b,
    float* __restrict__ proj)          // [N][384][HW] f32
{
  const int lane = threadIdx.x & 31;
  const int wave = threadIdx.x >> 5;
  const int n      = blockIdx.z;
  const int kblk   = blockIdx.y;
  const int hwbase = blockIdx.x * 64;

  const int krow = kblk * 64 + wave * 16 + (lane & 15);
  const __bf16* wrow = wbf + (size_t)krow * C_;
  const __bf16* xb = xT + (size_t)n * HW_ * C_;

  v8f acc[4] = {v8f{}, v8f{}, v8f{}, v8f{}};
  for (int kc = 0; kc < C_; kc += 32) {
    const v16bf a = load_a16(wrow + kc, lane);
    #pragma unroll
    for (int t = 0; t < 4; ++t) {
      const int col = hwbase + t * 16 + (lane & 15);
      const v16bf b = load_b16(xb + (size_t)col * C_ + kc, lane);
      acc[t] = wmma_bf16(a, b, acc[t]);
    }
  }

  float* po = proj + (size_t)n * MPROJ_ * HW_;
  #pragma unroll
  for (int t = 0; t < 4; ++t) {
    const int col = hwbase + t * 16 + (lane & 15);
    #pragma unroll
    for (int r = 0; r < 8; ++r) {
      const int k2 = kblk * 64 + wave * 16 + ((lane >> 4) << 3) + r;
      const float bias = (k2 < 64) ? theta_b[k2]
                       : (k2 < 128) ? phi_b[k2 - 64] : g_b[k2 - 128];
      po[(size_t)k2 * HW_ + col] = acc[t][r] + bias;
    }
  }
}

// ---------------------------------------------------------------------------
// Kernel 1b: theta -> bf16, q-major [N][4096][64] (A-operand friendly)
// ---------------------------------------------------------------------------
__global__ void k_theta_cvt(const float* __restrict__ proj,
                            __bf16* __restrict__ thetaT)
{
  const int idx = blockIdx.x * blockDim.x + threadIdx.x;
  if (idx >= N_ * HW_ * C8_) return;
  const int k = idx & 63;
  const int q = (idx >> 6) & (HW_ - 1);
  const int n = idx >> 18;
  thetaT[idx] = (__bf16)proj[((size_t)n * MPROJ_ + k) * HW_ + q];
}

// Kernel 1c: phi 2x2 maxpool -> bf16 [N][1024][64] p-major (energy-B friendly)
__global__ void k_pool_phi(const float* __restrict__ proj,
                           __bf16* __restrict__ phiT)
{
  const int idx = blockIdx.x * blockDim.x + threadIdx.x;
  if (idx >= N_ * HW4_ * C8_) return;
  const int k = idx & 63;
  const int p = (idx >> 6) & 1023;
  const int n = idx >> 16;
  const int h2 = p >> 5, w2 = p & 31;
  const float* s = proj + ((size_t)n * MPROJ_ + 64 + k) * HW_ + (h2 * 2) * W_ + w2 * 2;
  phiT[idx] = (__bf16)fmaxf(fmaxf(s[0], s[1]), fmaxf(s[W_], s[W_ + 1]));
}

// Kernel 1d: g 2x2 maxpool -> bf16 [N][256][1024] c-major (attn-B friendly)
__global__ void k_pool_g(const float* __restrict__ proj,
                         __bf16* __restrict__ gcm)
{
  const int idx = blockIdx.x * blockDim.x + threadIdx.x;
  if (idx >= N_ * C2_ * HW4_) return;
  const int p = idx & 1023;
  const int c = (idx >> 10) & 255;
  const int n = idx >> 18;
  const int h2 = p >> 5, w2 = p & 31;
  const float* s = proj + ((size_t)n * MPROJ_ + 128 + c) * HW_ + (h2 * 2) * W_ + w2 * 2;
  gcm[idx] = (__bf16)fmaxf(fmaxf(s[0], s[1]), fmaxf(s[W_], s[W_ + 1]));
}

// ---------------------------------------------------------------------------
// Kernel 2: fused attention.  One WG = 16 queries of one batch.
//   energy[16x1024] = theta[16x64] @ phi[64x1024]   (WMMA -> LDS f32)
//   softmax rows in LDS; probs written to LDS as bf16 in A layout
//   attnout[16x256] = attn[16x1024] @ g^T            (WMMA, A from LDS b128s)
// LDS: 16x1025 f32 + 16x1032 bf16 + reductions ~= 99 KB (< 320 KB/WGP).
// ---------------------------------------------------------------------------
__global__ __launch_bounds__(128) void k_attn(
    const __bf16* __restrict__ thetaT,   // [N][4096][64]
    const __bf16* __restrict__ phiT,     // [N][1024][64]
    const __bf16* __restrict__ gcm,      // [N][256][1024]
    __bf16* __restrict__ attnout)        // [N][4096][256]
{
  __shared__ float  sE[16][1025];
  __shared__ __bf16 sEb[16][1032];
  __shared__ float  sRed[16][8];
  __shared__ float  sRow[16];

  const int lane  = threadIdx.x & 31;
  const int wave  = threadIdx.x >> 5;
  const int n     = blockIdx.y;
  const int qbase = blockIdx.x * 16;

  // ---- phase 1: energy tile -> LDS ----
  const __bf16* th = thetaT + ((size_t)n * HW_ + qbase) * C8_;
  const v16bf a0 = load_a16(th + (size_t)(lane & 15) * C8_, lane);
  const v16bf a1 = load_a16(th + (size_t)(lane & 15) * C8_ + 32, lane);

  const __bf16* ph = phiT + (size_t)n * HW4_ * C8_;
  const int pbase = wave * 256;
  for (int t = 0; t < 16; ++t) {
    const int p = pbase + t * 16 + (lane & 15);
    const __bf16* pb = ph + (size_t)p * C8_;
    const v16bf b0 = load_b16(pb, lane);
    const v16bf b1 = load_b16(pb + 32, lane);
    v8f acc = {};
    acc = wmma_bf16(a0, b0, acc);
    acc = wmma_bf16(a1, b1, acc);
    const int mrow = (lane >> 4) << 3;
    #pragma unroll
    for (int r = 0; r < 8; ++r)
      sE[mrow + r][p] = acc[r];
  }
  __syncthreads();

  // ---- phase 2: softmax (8 threads per row); probs -> sEb (bf16) ----
  {
    const int row = threadIdx.x >> 3;
    const int sub = threadIdx.x & 7;
    float lmax = -3.0e38f;
    for (int j = sub; j < 1024; j += 8) lmax = fmaxf(lmax, sE[row][j]);
    sRed[row][sub] = lmax;
    __syncthreads();
    float rmax = sRed[row][0];
    #pragma unroll
    for (int i = 1; i < 8; ++i) rmax = fmaxf(rmax, sRed[row][i]);
    float lsum = 0.f;
    for (int j = sub; j < 1024; j += 8) {
      const float ev = __expf(sE[row][j] - rmax);
      sE[row][j] = ev;
      lsum += ev;
    }
    __syncthreads();
    sRed[row][sub] = lsum;
    __syncthreads();
    if (sub == 0) {
      float ssum = 0.f;
      #pragma unroll
      for (int i = 0; i < 8; ++i) ssum += sRed[row][i];
      sRow[row] = 1.0f / ssum;
    }
    __syncthreads();
    const float inv = sRow[row];
    for (int j = sub; j < 1024; j += 8)
      sEb[row][j] = (__bf16)(sE[row][j] * inv);
  }
  __syncthreads();

  // ---- phase 3: attn @ g^T -> attnout ----
  const __bf16* gb = gcm + (size_t)n * C2_ * HW4_;
  const int cbase = wave * 64;
  v8f acc[4] = {v8f{}, v8f{}, v8f{}, v8f{}};
  for (int pc = 0; pc < 1024; pc += 32) {
    const v16bf a = load_a16(&sEb[lane & 15][pc], lane);
    #pragma unroll
    for (int t = 0; t < 4; ++t) {
      const int c = cbase + t * 16 + (lane & 15);
      const v16bf b = load_b16(gb + (size_t)c * HW4_ + pc, lane);
      acc[t] = wmma_bf16(a, b, acc[t]);
    }
  }
  __bf16* ao = attnout + ((size_t)n * HW_ + qbase) * C2_;
  #pragma unroll
  for (int t = 0; t < 4; ++t) {
    const int c = cbase + t * 16 + (lane & 15);
    #pragma unroll
    for (int r = 0; r < 8; ++r)
      ao[(size_t)(((lane >> 4) << 3) + r) * C2_ + c] = (__bf16)acc[t][r];
  }
}

// ---------------------------------------------------------------------------
// Kernel 3: final 1x1 conv + bias + gamma*out + residual.
// M=512 (k), N=4096 (q), K=256 (c). WG 128thr: 64k x 64q tile.
// ---------------------------------------------------------------------------
__global__ __launch_bounds__(128) void k_final(
    const __bf16* __restrict__ attnout, // [N][4096][256]
    const __bf16* __restrict__ owbf,    // [512][256] bf16
    const float* __restrict__ out_b,    // [512]
    const float* __restrict__ gamma,    // [1]
    const float* __restrict__ x,        // [N][512][4096]
    float* __restrict__ out)            // [N][512][4096]
{
  const int lane  = threadIdx.x & 31;
  const int wave  = threadIdx.x >> 5;
  const int n     = blockIdx.z;
  const int kbase = blockIdx.y * 64 + wave * 16;
  const int qbase = blockIdx.x * 64;
  const float gm  = gamma[0];

  const __bf16* wr = owbf + (size_t)(kbase + (lane & 15)) * C2_;
  const __bf16* ab = attnout + (size_t)n * HW_ * C2_;

  v8f acc[4] = {v8f{}, v8f{}, v8f{}, v8f{}};
  for (int cc = 0; cc < C2_; cc += 32) {
    const v16bf a = load_a16(wr + cc, lane);
    #pragma unroll
    for (int t = 0; t < 4; ++t) {
      const int q = qbase + t * 16 + (lane & 15);
      const v16bf b = load_b16(ab + (size_t)q * C2_ + cc, lane);
      acc[t] = wmma_bf16(a, b, acc[t]);
    }
  }
  #pragma unroll
  for (int t = 0; t < 4; ++t) {
    const int q = qbase + t * 16 + (lane & 15);
    #pragma unroll
    for (int r = 0; r < 8; ++r) {
      const int k = kbase + ((lane >> 4) << 3) + r;
      const size_t oi = ((size_t)n * C_ + k) * HW_ + q;
      out[oi] = gm * (acc[t][r] + out_b[k]) + x[oi];
    }
  }
}

// ---------------------------------------------------------------------------
extern "C" void kernel_launch(void* const* d_in, const int* in_sizes, int n_in,
                              void* d_out, int out_size, void* d_ws, size_t ws_size,
                              hipStream_t stream)
{
  const float* x       = (const float*)d_in[0];
  const float* theta_w = (const float*)d_in[1];
  const float* theta_b = (const float*)d_in[2];
  const float* phi_w   = (const float*)d_in[3];
  const float* phi_b   = (const float*)d_in[4];
  const float* g_w     = (const float*)d_in[5];
  const float* g_b     = (const float*)d_in[6];
  const float* out_w   = (const float*)d_in[7];
  const float* out_b   = (const float*)d_in[8];
  const float* gamma   = (const float*)d_in[9];

  char* ws = (char*)d_ws;
  size_t off = 0;
  auto carve = [&](size_t bytes) -> void* {
    void* p = ws + off;
    off = (off + bytes + 255) & ~(size_t)255;
    return p;
  };
  float*  proj    = (float*) carve((size_t)N_ * MPROJ_ * HW_ * sizeof(float)); // 50.3 MB
  __bf16* xT      = (__bf16*)carve((size_t)N_ * HW_  * C_  * 2);               // 33.6 MB
  __bf16* wbf     = (__bf16*)carve((size_t)MPROJ_ * C_ * 2);                   //  0.4 MB
  __bf16* owbf    = (__bf16*)carve((size_t)C_ * C2_ * 2);                      //  0.3 MB
  __bf16* thetaT  = (__bf16*)carve((size_t)N_ * HW_  * C8_ * 2);               //  4.2 MB
  __bf16* phiT    = (__bf16*)carve((size_t)N_ * HW4_ * C8_ * 2);               //  1.0 MB
  __bf16* gcm     = (__bf16*)carve((size_t)N_ * C2_  * HW4_ * 2);              //  4.2 MB
  __bf16* attnout = (__bf16*)carve((size_t)N_ * HW_  * C2_ * 2);               // 16.8 MB

  { const int tot = MPROJ_ * C_ + C_ * C2_;
    k_cvt_w<<<(tot + 255) / 256, 256, 0, stream>>>(theta_w, phi_w, g_w, out_w,
                                                   wbf, owbf); }
  k_xpose<<<dim3(HW_ / 32, C_ / 32, N_), 256, 0, stream>>>(x, xT);

  k_proj<<<dim3(HW_ / 64, MPROJ_ / 64, N_), 128, 0, stream>>>(
      xT, wbf, theta_b, phi_b, g_b, proj);

  { const int tot = N_ * HW_ * C8_;
    k_theta_cvt<<<(tot + 255) / 256, 256, 0, stream>>>(proj, thetaT); }
  { const int tot = N_ * HW4_ * C8_;
    k_pool_phi<<<(tot + 255) / 256, 256, 0, stream>>>(proj, phiT); }
  { const int tot = N_ * C2_ * HW4_;
    k_pool_g<<<(tot + 255) / 256, 256, 0, stream>>>(proj, gcm); }

  k_attn<<<dim3(HW_ / 16, N_), 128, 0, stream>>>(thetaT, phiT, gcm, attnout);

  k_final<<<dim3(HW_ / 64, C_ / 64, N_), 128, 0, stream>>>(
      attnout, owbf, out_b, gamma, x, (float*)d_out);
}
